// BahdanauAttention_14740327760513
// MI455X (gfx1250) — matrix operand order repaired and checked by reading, not compiled, not run on hardware
//
#include <hip/hip_runtime.h>
#include <hip/hip_bf16.h>
#include <stdint.h>

// ---------------------------------------------------------------------------
// Bahdanau attention, B=8, L=512, D=128 on MI455X (gfx1250, wave32, WMMA+TDM).
//  C1: one-time f32 -> bf16 packed copies (H, S, weights)
//  C2: one-time bf16 transpose HT[b][d][k] (LDS-tiled, coalesced both sides)
//  K0: Wh = H @ Wh_w^T, Ws = S @ Ws_w^T   (bf16 WMMA, f32 accum, 32 wmma/wave)
//  K1: fused per (b, 16-row block): scores -> softmax -> P @ H  (bf16 WMMA)
//      phase-1 Ws staging via Tensor Data Mover, double-buffered in LDS.
// ---------------------------------------------------------------------------

#define B_ 8
#define L_ 512
#define D_ 128

typedef __attribute__((ext_vector_type(16))) __bf16    v16bf;
typedef __attribute__((ext_vector_type(8)))  __bf16    v8bf;
typedef __attribute__((ext_vector_type(8)))  float     v8f;
typedef __attribute__((ext_vector_type(4)))  unsigned  v4u;
typedef __attribute__((ext_vector_type(8)))  int       v8i;
typedef __attribute__((ext_vector_type(4)))  int       v4i;

#if __has_builtin(__builtin_amdgcn_tensor_load_to_lds) && \
    __has_builtin(__builtin_amdgcn_s_wait_tensorcnt)
#define USE_TDM 1
#else
#define USE_TDM 0
#endif

static __device__ __forceinline__ unsigned bfbits(float f) {  // RNE f32->bf16
    unsigned u = __builtin_bit_cast(unsigned, f);
    return (u + 0x7FFFu + ((u >> 16) & 1u)) >> 16;
}
static __device__ __forceinline__ unsigned pack2bf(float x, float y) {
    return bfbits(x) | (bfbits(y) << 16);
}
static __device__ __forceinline__ float tanh_fast(float x) {
#if __has_builtin(__builtin_amdgcn_tanhf)
    return __builtin_amdgcn_tanhf(x);          // v_tanh_f32 (CDNA5 trans op)
#else
    float y;                                   // raw CDNA5 trans op + hazard nop
    asm volatile("v_tanh_f32 %0, %1\n\tv_nop" : "=v"(y) : "v"(x));
    return y;
#endif
}

#if USE_TDM
// Issue one TDM load: 16 rows x 128 f32 from Ws into LDS at lds_off, with
// hardware row padding of 4 DWORDs (pitch 132 f32).  One call per wave.
static __device__ __forceinline__ void tdm_load_ws(const float* gsrc, unsigned lds_off) {
    const unsigned long long ga = (unsigned long long)(const void*)gsrc;
    const v4u g0 = { 1u,                                  // count=1, user desc
                     lds_off,                             // LDS byte address
                     (unsigned)(ga & 0xFFFFFFFFull),
                     (unsigned)(((ga >> 32) & 0x1FFFFFFull) | (2u << 30)) }; // type=2
    // data_size=4B(2)<<16 | pad_enable<<20 | pad_interval=6(128DW)<<22 | pad_amount=3(4DW)<<25
    const v8i g1 = { (int)0x07920000u,
                     (int)(128u << 16),                   // tensor_dim0 = 128
                     (int)(512u << 16),                   // tensor_dim1 = 512
                     (int)(128u << 16),                   // tile_dim0  = 128
                     16,                                  // tile_dim1  = 16
                     128,                                 // tensor_dim0_stride = 128
                     0, 0 };
    const v4i g2 = { 0, 0, 0, 0 };
    const v4i g3 = { 0, 0, 0, 0 };
#if __clang_major__ >= 23
    const v8i g4 = { 0, 0, 0, 0, 0, 0, 0, 0 };
    __builtin_amdgcn_tensor_load_to_lds(g0, g1, g2, g3, g4, 0);
#else
    __builtin_amdgcn_tensor_load_to_lds(g0, g1, g2, g3, 0);
#endif
}
#endif

// --------------------------- C1: pairwise convert ---------------------------
__global__ __launch_bounds__(256)
void cvt_pairs(const float* __restrict__ src, unsigned* __restrict__ dst, int npairs) {
    const int i = blockIdx.x * 256 + threadIdx.x;
    if (i < npairs) {
        const float2 p = ((const float2*)src)[i];
        dst[i] = pack2bf(p.x, p.y);
    }
}

// ------------------- C2: HT[b][d][k] = bf16(H[b][k][d]) ---------------------
__global__ __launch_bounds__(256)
void transpose_bf16(const float* __restrict__ H, __bf16* __restrict__ HT) {
    __shared__ __bf16 tb[128][40];             // pitch 80B (16B aligned rows)
    const int b = blockIdx.x, kc = blockIdx.y; // 16 chunks of 32 k
    const int tid = threadIdx.x;
    const int kbase = kc * 32;
    #pragma unroll
    for (int i = 0; i < 16; ++i) {             // 4096 elems, coalesced reads
        const int idx = tid + i * 256;
        const int k = idx >> 7, d = idx & 127;
        const float v = H[((size_t)b * L_ + kbase + k) * D_ + d];
        tb[d][k] = __builtin_bit_cast(__bf16, (unsigned short)bfbits(v));
    }
    __syncthreads();
    const int d = tid >> 1, part = tid & 1;    // 32B contiguous per thread
    const uint4* s = (const uint4*)&tb[d][part * 16];
    uint4* o = (uint4*)(HT + ((size_t)b * D_ + d) * L_ + kbase + part * 16);
    o[0] = s[0];
    o[1] = s[1];
}

// ----------------- K0: projections, one 16x128 stripe per wave --------------
__global__ __launch_bounds__(256)
void proj_gemm(const __bf16* __restrict__ Hbf, const __bf16* __restrict__ Sbf,
               const __bf16* __restrict__ WhWb, const __bf16* __restrict__ WsWb,
               float* __restrict__ WhBuf, float* __restrict__ WsBuf) {
    const int wave = threadIdx.x >> 5, lane = threadIdx.x & 31;
    const int st  = blockIdx.x * 8 + wave;     // 0..511 stripes
    const int mat = st >> 8;                   // 0: H path, 1: S path
    const int tm  = st & 255;
    const __bf16* X = mat ? Sbf  : Hbf;
    const __bf16* W = mat ? WsWb : WhWb;
    float*        Y = mat ? WsBuf : WhBuf;
    const int r = lane & 15, hi = lane >> 4;

    const __bf16* arow = X + (size_t)(tm * 16 + r) * D_;
    v8f acc[8] = {};
    #pragma unroll
    for (int k0 = 0; k0 < D_; k0 += 32) {
        const v8bf alo = *(const v8bf*)(arow + k0 + hi * 16);
        const v8bf ahi = *(const v8bf*)(arow + k0 + hi * 16 + 8);
        v16bf a;
        #pragma unroll
        for (int v = 0; v < 8; ++v) { a[v] = alo[v]; a[8 + v] = ahi[v]; }
        #pragma unroll
        for (int tn = 0; tn < 8; ++tn) {       // A-fragment reused 8x
            const v16bf bb = *(const v16bf*)(W + (size_t)(tn * 16 + r) * D_ + k0 + hi * 16);
            acc[tn] = __builtin_amdgcn_wmma_f32_16x16x32_bf16(false, a, false, bb,
                                                              (short)0, acc[tn], false, false);
        }
    }
    #pragma unroll
    for (int tn = 0; tn < 8; ++tn) {
        float* yp = Y + (size_t)(tm * 16 + hi * 8) * D_ + tn * 16 + r;
        #pragma unroll
        for (int j = 0; j < 8; ++j) yp[(size_t)j * D_] = acc[tn][j];
    }
}

// ------------- K1: fused scores + softmax + P@H per 16-row block ------------
// smem byte layout (single __shared__ block => LDS offset 0):
//   [0,33024)       sc   : f32 [16][516] raw score strip
//   [33024,58880)   phase1: wh f32[16][132] | ws f32[2][16][132] | vsh f32[128]
//                   phase3: scb bf16[16][536] (aliased, 17152B)
//   [58880,59968)   red  : f32 [16][17]
//   [59968,60096)   rowMax[16], rowSum[16]
#define SC_OFF   0
#define WH_OFF   33024
#define WS_OFF   (33024 + 8448)
#define VSH_OFF  (33024 + 8448 + 16896)
#define RED_OFF  58880
#define RMAX_OFF 59968
#define WS_BUF_BYTES 8448

__global__ __launch_bounds__(256)
void attn_fused(const uint8_t* __restrict__ mask,
                const float* __restrict__ Wh, const float* __restrict__ Ws,
                const float* __restrict__ Vw, const __bf16* __restrict__ HT,
                float* __restrict__ out) {
    __shared__ __align__(16) char smem[60096];
    float  (*sc)[516]      = (float (*)[516])(smem + SC_OFF);
    float  (*wh)[132]      = (float (*)[132])(smem + WH_OFF);
    float  (*ws)[16][132]  = (float (*)[16][132])(smem + WS_OFF);
    float*  vsh            = (float*)(smem + VSH_OFF);
    __bf16 (*scb)[536]     = (__bf16 (*)[536])(smem + WH_OFF);   // aliases phase-1
    float  (*red)[17]      = (float (*)[17])(smem + RED_OFF);
    float*  rowMax         = (float*)(smem + RMAX_OFF);
    float*  rowSum         = rowMax + 16;

    const int b   = blockIdx.x;                // 0..7
    const int lb  = blockIdx.y;                // 0..31
    const int tid = threadIdx.x;
    const int wv  = tid >> 5;

    if (tid < 128) vsh[tid] = Vw[tid];
    for (int i = tid; i < 16 * 128; i += 256) {
        const int rr = i >> 7, cc = i & 127;
        wh[rr][cc] = Wh[((size_t)(b * L_ + lb * 16 + rr)) * D_ + cc];
    }

    const int l  = tid >> 4;                   // 0..15 query row
    const int mi = tid & 15;                   // 0..15 key within chunk
    const int lg = lb * 16 + l;
    const uint8_t* mrow = mask + ((size_t)b * L_ + lg) * L_;
    const float4* vshv = (const float4*)vsh;
    const float* wsrc = Ws + ((size_t)b * L_) * D_;

    // ---- phase 1: 16x512 score strip; Ws chunks staged by the TDM ----------
#if USE_TDM
    if (wv == 0) tdm_load_ws(wsrc, WS_OFF);            // chunk 0 -> buf 0
    for (int i = 0; i < 32; ++i) {
        const int m0 = i * 16;
        if (wv == 0) {
            if (i + 1 < 32) {                          // pipeline next chunk
                tdm_load_ws(wsrc + (size_t)(m0 + 16) * D_,
                            WS_OFF + ((unsigned)((i + 1) & 1)) * WS_BUF_BYTES);
                __builtin_amdgcn_s_wait_tensorcnt(1);  // chunk i retired
            } else {
                __builtin_amdgcn_s_wait_tensorcnt(0);
            }
        }
        __syncthreads();
        const float4* whv = (const float4*)&wh[l][0];
        const float4* wsv = (const float4*)&ws[i & 1][mi][0];
        float s = 0.0f;
        #pragma unroll 4
        for (int d4 = 0; d4 < 32; ++d4) {
            const float4 a4 = whv[d4], b4 = wsv[d4], v4 = vshv[d4];
            s = fmaf(v4.x, tanh_fast(a4.x + b4.x), s);
            s = fmaf(v4.y, tanh_fast(a4.y + b4.y), s);
            s = fmaf(v4.z, tanh_fast(a4.z + b4.z), s);
            s = fmaf(v4.w, tanh_fast(a4.w + b4.w), s);
        }
        sc[l][m0 + mi] = mrow[m0 + mi] ? -__builtin_inff() : s;
        __syncthreads();
    }
#else
    __syncthreads();
    for (int i = 0; i < 32; ++i) {
        const int m0 = i * 16;
        for (int j = tid; j < 16 * 128; j += 256) {
            const int rr = j >> 7, cc = j & 127;
            ws[i & 1][rr][cc] = wsrc[(size_t)(m0 + rr) * D_ + cc];
        }
        if (m0 + 16 < L_)
            __builtin_prefetch(&wsrc[(size_t)(m0 + 16 + l) * D_ + mi * 8], 0, 3);
        __syncthreads();
        const float4* whv = (const float4*)&wh[l][0];
        const float4* wsv = (const float4*)&ws[i & 1][mi][0];
        float s = 0.0f;
        #pragma unroll 4
        for (int d4 = 0; d4 < 32; ++d4) {
            const float4 a4 = whv[d4], b4 = wsv[d4], v4 = vshv[d4];
            s = fmaf(v4.x, tanh_fast(a4.x + b4.x), s);
            s = fmaf(v4.y, tanh_fast(a4.y + b4.y), s);
            s = fmaf(v4.z, tanh_fast(a4.z + b4.z), s);
            s = fmaf(v4.w, tanh_fast(a4.w + b4.w), s);
        }
        sc[l][m0 + mi] = mrow[m0 + mi] ? -__builtin_inff() : s;
        __syncthreads();
    }
#endif

    // ---- phase 2: softmax over 512 keys; emit packed bf16 probabilities ----
    float vals[32];
    {
        const float4* srow = (const float4*)&sc[l][mi * 32];
        #pragma unroll
        for (int q = 0; q < 8; ++q) {
            const float4 t = srow[q];
            vals[4 * q + 0] = t.x; vals[4 * q + 1] = t.y;
            vals[4 * q + 2] = t.z; vals[4 * q + 3] = t.w;
        }
    }
    float mx = -__builtin_inff();
    #pragma unroll
    for (int j = 0; j < 32; ++j) mx = fmaxf(mx, vals[j]);
    red[l][mi] = mx;
    __syncthreads();
    if (mi == 0) {
        float m2 = red[l][0];
        #pragma unroll
        for (int j = 1; j < 16; ++j) m2 = fmaxf(m2, red[l][j]);
        rowMax[l] = m2;
    }
    __syncthreads();
    const float rm = rowMax[l];
    float sum = 0.0f;
    #pragma unroll
    for (int j = 0; j < 32; ++j) { vals[j] = __expf(vals[j] - rm); sum += vals[j]; }
    red[l][mi] = sum;
    __syncthreads();
    if (mi == 0) {
        float s2 = 0.0f;
        #pragma unroll
        for (int j = 0; j < 16; ++j) s2 += red[l][j];
        rowSum[l] = s2;
    }
    __syncthreads();                            // wh/ws/vsh dead after here
    const float inv = 1.0f / rowSum[l];
    {
        unsigned* dst = (unsigned*)&scb[l][mi * 32];
        #pragma unroll
        for (int q = 0; q < 16; ++q)
            dst[q] = pack2bf(vals[2 * q] * inv, vals[2 * q + 1] * inv);
    }
    __syncthreads();

    // ---- phase 3: rSeq(16x128) = P(16x512) @ H_b via bf16 WMMA -------------
    const int lane = tid & 31;
    const int r = lane & 15, hi = lane >> 4;
    const __bf16* HTb = HT + ((size_t)b * D_ + wv * 16 + r) * L_;

    v8f c = {};
    #pragma unroll 4
    for (int k0 = 0; k0 < L_; k0 += 32) {
        const v8bf alo = *(const v8bf*)&scb[r][k0 + hi * 16];
        const v8bf ahi = *(const v8bf*)&scb[r][k0 + hi * 16 + 8];
        v16bf a;
        #pragma unroll
        for (int v = 0; v < 8; ++v) { a[v] = alo[v]; a[8 + v] = ahi[v]; }
        const v16bf bb = *(const v16bf*)(HTb + k0 + hi * 16);   // 32B contiguous
        c = __builtin_amdgcn_wmma_f32_16x16x32_bf16(false, a, false, bb,
                                                    (short)0, c, false, false);
    }
    float* orow = out + ((size_t)(b * L_ + lb * 16 + hi * 8)) * D_ + wv * 16 + r;
    #pragma unroll
    for (int j = 0; j < 8; ++j) orow[(size_t)j * D_] = c[j];
}

// ---------------------------------------------------------------------------
extern "C" void kernel_launch(void* const* d_in, const int* in_sizes, int n_in,
                              void* d_out, int out_size, void* d_ws, size_t ws_size,
                              hipStream_t stream) {
    const float*   H    = (const float*)d_in[0];
    const float*   S    = (const float*)d_in[1];
    const uint8_t* mask = (const uint8_t*)d_in[2];   // jnp.bool_: 1 byte/elem
    const float*   WhW  = (const float*)d_in[3];
    const float*   WsW  = (const float*)d_in[4];
    const float*   Vw   = (const float*)d_in[5];
    float*         out  = (float*)d_out;

    char* wsb = (char*)d_ws;
    const size_t BLD = (size_t)B_ * L_ * D_;         // 524288
    float*  WhBuf = (float*)(wsb);                   // 2 MB
    float*  WsBuf = (float*)(wsb + 2097152);         // 2 MB
    __bf16* Hbf   = (__bf16*)(wsb + 4194304);        // 1 MB
    __bf16* Sbf   = (__bf16*)(wsb + 5242880);        // 1 MB
    __bf16* HTbf  = (__bf16*)(wsb + 6291456);        // 1 MB
    __bf16* WhWb  = (__bf16*)(wsb + 7340032);        // 32 KB
    __bf16* WsWb  = (__bf16*)(wsb + 7372800);        // 32 KB

    cvt_pairs<<<(int)(BLD / 2 / 256), 256, 0, stream>>>(H,   (unsigned*)Hbf,  (int)(BLD / 2));
    cvt_pairs<<<(int)(BLD / 2 / 256), 256, 0, stream>>>(S,   (unsigned*)Sbf,  (int)(BLD / 2));
    cvt_pairs<<<32, 256, 0, stream>>>(WhW, (unsigned*)WhWb, D_ * D_ / 2);
    cvt_pairs<<<32, 256, 0, stream>>>(WsW, (unsigned*)WsWb, D_ * D_ / 2);
    transpose_bf16<<<dim3(B_, 16), 256, 0, stream>>>(H, HTbf);

    proj_gemm<<<64, 256, 0, stream>>>(Hbf, Sbf, WhWb, WsWb, WhBuf, WsBuf);
    attn_fused<<<dim3(B_, L_ / 16), 256, 0, stream>>>(mask, WhBuf, WsBuf, Vw, HTbf, out);
}
